// GCL_86620900426032
// MI455X (gfx1250) — matrix operand order repaired and checked by reading, not compiled, not run on hardware
//
#include <hip/hip_runtime.h>

#define D 128

typedef __attribute__((ext_vector_type(16))) __bf16 bf16x16;
typedef __attribute__((ext_vector_type(8)))  __bf16 bf16x8;
typedef __attribute__((ext_vector_type(8)))  float  f32x8;

// round-to-nearest-even f32 -> bf16 bit pattern
__device__ __forceinline__ unsigned short f2bf(float f) {
  unsigned int u = __float_as_uint(f);
  u += 0x7FFFu + ((u >> 16) & 1u);
  return (unsigned short)(u >> 16);
}

// Assemble a 16x32 bf16 A-fragment for this lane from two contiguous
// 16-byte LDS chunks (ISA 16-bit A layout: lane<16 -> K {0..7,16..23},
// lane>=16 -> K {8..15,24..31}; caller passes the right base).
__device__ __forceinline__ bf16x16 load_a_frag(const unsigned short* p0,
                                               const unsigned short* p1) {
  bf16x8 lo = *(const bf16x8*)p0;
  bf16x8 hi = *(const bf16x8*)p1;
  bf16x16 a;
#pragma unroll
  for (int i = 0; i < 8; ++i) { a[i] = lo[i]; a[i + 8] = hi[i]; }
  return a;
}

__device__ __forceinline__ f32x8 wmma_bf16(bf16x16 a, bf16x16 b, f32x8 c) {
  // (neg_a, A, neg_b, B, c_mod, C, reuse_a, reuse_b)
  return __builtin_amdgcn_wmma_f32_16x16x32_bf16(false, a, false, b,
                                                 (short)0, c, false, false);
}

// ---------------------------------------------------------------------------
// Prep: pack fp32 weights [K,128] into per-lane WMMA B-fragment layout:
// packed[((kt*8+nt)*32+lane)*16 + i] = bf16(W[kt*32 + (lane>>4)*8 + (i&7) + (i>>3)*16][nt*16 + (lane&15)])
// so the GEMM loop loads one contiguous 32B run per lane per fragment.
// ---------------------------------------------------------------------------
__global__ void pack_w_kernel(const float* __restrict__ W,
                              unsigned short* __restrict__ out, int KT) {
  int t = blockIdx.x * blockDim.x + threadIdx.x;
  int total = KT * 8 * 32 * 16;
  if (t >= total) return;
  int i    = t & 15;
  int lane = (t >> 4) & 31;
  int nt   = (t >> 9) & 7;
  int kt   = t >> 12;
  int k = kt * 32 + ((lane >> 4) << 3) + (i & 7) + ((i >> 3) << 4);
  int n = nt * 16 + (lane & 15);
  out[t] = f2bf(W[k * D + n]);
}

// Convert x to bf16 (halves gather bandwidth) and zero the aggregation buffer.
__global__ void cvt_zero_kernel(const float* __restrict__ x,
                                unsigned short* __restrict__ xb,
                                float* __restrict__ agg, long long total) {
  long long t = (long long)blockIdx.x * blockDim.x + threadIdx.x;
  if (t < total) { xb[t] = f2bf(x[t]); agg[t] = 0.0f; }
}

// ---------------------------------------------------------------------------
// Edge kernel: per wave, 16 edges. Gather concat(x[row],x[col]) -> LDS,
// layer1 (K=256) WMMA, ReLU, layer2 (K=128) WMMA, ReLU*mask,
// write edge_feat + atomic scatter-add into agg[row].
// ---------------------------------------------------------------------------
__global__ __launch_bounds__(128, 2) void egcl_edge_kernel(
    const int* __restrict__ ei, const float* __restrict__ emask,
    const unsigned short* __restrict__ xb,
    const unsigned short* __restrict__ w1, const float* __restrict__ b1,
    const unsigned short* __restrict__ w2, const float* __restrict__ b2,
    float* __restrict__ edge_out, float* __restrict__ agg, int E_) {
  __shared__ alignas(64) unsigned short sh[4][16 * 256];  // 32 KB
  const int wave = threadIdx.x >> 5;
  const int lane = threadIdx.x & 31;
  unsigned short* my = sh[wave];
  const long long e0 = (long long)blockIdx.x * 64 + wave * 16;

  const int cit   = lane & 15;         // column (and A-row) index in tile
  const int ksel  = (lane >> 4) << 3;  // K sub-select for A fragments
  const int mbase = (lane >> 4) << 3;  // C rows covered: mbase..mbase+7

  // ---- gather: edge t gets 32 x 16B chunks (16 from row node, 16 from col)
#pragma unroll
  for (int t = 0; t < 16; ++t) {
    long long eg = e0 + t;
    if (eg >= E_) eg = E_ - 1;
    int node = (lane < 16) ? ei[eg] : ei[(long long)E_ + eg];
    const uint4* src = (const uint4*)(xb + (long long)node * D + (lane & 15) * 8);
    *(uint4*)(my + t * 256 + lane * 8) = *src;
  }
  __syncthreads();

  // ---- layer 1: [16,256] @ [256,128]
  f32x8 acc[8];
#pragma unroll
  for (int nt = 0; nt < 8; ++nt) {
    float bv = b1[nt * 16 + cit];
    f32x8 c;
#pragma unroll
    for (int r = 0; r < 8; ++r) c[r] = bv;
    acc[nt] = c;
  }
#pragma unroll
  for (int kt = 0; kt < 8; ++kt) {
    const unsigned short* ap = my + cit * 256 + kt * 32 + ksel;
    bf16x16 a = load_a_frag(ap, ap + 16);
#pragma unroll
    for (int nt = 0; nt < 8; ++nt) {
      bf16x16 b = *(const bf16x16*)(w1 + (((kt << 3) + nt) * 32 + lane) * 16);
      acc[nt] = wmma_bf16(a, b, acc[nt]);
    }
  }

  // ---- ReLU, bf16, stage h [16,128] into this wave's LDS region
#pragma unroll
  for (int nt = 0; nt < 8; ++nt)
#pragma unroll
    for (int r = 0; r < 8; ++r) {
      float v = acc[nt][r];
      v = v > 0.0f ? v : 0.0f;
      my[(mbase + r) * D + nt * 16 + cit] = f2bf(v);
    }
  __syncthreads();

  // ---- layer 2: [16,128] @ [128,128]
  f32x8 acc2[8];
#pragma unroll
  for (int nt = 0; nt < 8; ++nt) {
    float bv = b2[nt * 16 + cit];
    f32x8 c;
#pragma unroll
    for (int r = 0; r < 8; ++r) c[r] = bv;
    acc2[nt] = c;
  }
#pragma unroll
  for (int kt = 0; kt < 4; ++kt) {
    const unsigned short* ap = my + cit * D + kt * 32 + ksel;
    bf16x16 a = load_a_frag(ap, ap + 16);
#pragma unroll
    for (int nt = 0; nt < 8; ++nt) {
      bf16x16 b = *(const bf16x16*)(w2 + (((kt << 3) + nt) * 32 + lane) * 16);
      acc2[nt] = wmma_bf16(a, b, acc2[nt]);
    }
  }

  // ---- epilogue: ReLU * mask, write edge_feat, scatter-add to agg[row]
  float mk[8];
  int rw[8];
#pragma unroll
  for (int r = 0; r < 8; ++r) {
    long long e = e0 + mbase + r;
    long long ec = e < E_ ? e : E_ - 1;
    mk[r] = emask[ec];
    rw[r] = ei[ec];
  }
#pragma unroll
  for (int nt = 0; nt < 8; ++nt) {
    int c = nt * 16 + cit;
#pragma unroll
    for (int r = 0; r < 8; ++r) {
      long long e = e0 + mbase + r;
      if (e < E_) {
        float v = acc2[nt][r];
        v = v > 0.0f ? v : 0.0f;
        v *= mk[r];
        edge_out[e * D + c] = v;
        unsafeAtomicAdd(agg + (long long)rw[r] * D + c, v);  // L2-resident
      }
    }
  }
}

// ---------------------------------------------------------------------------
// Node kernel: per wave, 16 nodes. A = concat(x_bf16[n], bf16(agg[n])),
// layer1 WMMA + ReLU, layer2 WMMA (no ReLU), add bias + residual x.
// ---------------------------------------------------------------------------
__global__ __launch_bounds__(128, 2) void egcl_node_kernel(
    const float* __restrict__ x, const unsigned short* __restrict__ xb,
    const float* __restrict__ agg,
    const unsigned short* __restrict__ w1, const float* __restrict__ b1,
    const unsigned short* __restrict__ w2, const float* __restrict__ b2,
    float* __restrict__ out, int N_) {
  __shared__ alignas(64) unsigned short sh[4][16 * 256];  // 32 KB
  const int wave = threadIdx.x >> 5;
  const int lane = threadIdx.x & 31;
  unsigned short* my = sh[wave];
  const long long n0 = (long long)blockIdx.x * 64 + wave * 16;

  const int cit   = lane & 15;
  const int ksel  = (lane >> 4) << 3;
  const int mbase = (lane >> 4) << 3;

  // ---- gather: cols 0..127 from x_bf16, cols 128..255 from agg (fp32->bf16)
#pragma unroll
  for (int t = 0; t < 16; ++t) {
    long long n = n0 + t;
    if (n >= N_) n = N_ - 1;
    if (lane < 16) {
      *(uint4*)(my + t * 256 + lane * 8) =
          *(const uint4*)(xb + n * D + lane * 8);
    } else {
      const float* s = agg + n * D + (lane & 15) * 8;
      uint4 v;
      v.x = (unsigned)f2bf(s[0]) | ((unsigned)f2bf(s[1]) << 16);
      v.y = (unsigned)f2bf(s[2]) | ((unsigned)f2bf(s[3]) << 16);
      v.z = (unsigned)f2bf(s[4]) | ((unsigned)f2bf(s[5]) << 16);
      v.w = (unsigned)f2bf(s[6]) | ((unsigned)f2bf(s[7]) << 16);
      *(uint4*)(my + t * 256 + lane * 8) = v;
    }
  }
  __syncthreads();

  // ---- layer 1
  f32x8 acc[8];
#pragma unroll
  for (int nt = 0; nt < 8; ++nt) {
    float bv = b1[nt * 16 + cit];
    f32x8 c;
#pragma unroll
    for (int r = 0; r < 8; ++r) c[r] = bv;
    acc[nt] = c;
  }
#pragma unroll
  for (int kt = 0; kt < 8; ++kt) {
    const unsigned short* ap = my + cit * 256 + kt * 32 + ksel;
    bf16x16 a = load_a_frag(ap, ap + 16);
#pragma unroll
    for (int nt = 0; nt < 8; ++nt) {
      bf16x16 b = *(const bf16x16*)(w1 + (((kt << 3) + nt) * 32 + lane) * 16);
      acc[nt] = wmma_bf16(a, b, acc[nt]);
    }
  }

  // ---- ReLU, stage h
#pragma unroll
  for (int nt = 0; nt < 8; ++nt)
#pragma unroll
    for (int r = 0; r < 8; ++r) {
      float v = acc[nt][r];
      v = v > 0.0f ? v : 0.0f;
      my[(mbase + r) * D + nt * 16 + cit] = f2bf(v);
    }
  __syncthreads();

  // ---- layer 2 (no ReLU)
  f32x8 acc2[8];
#pragma unroll
  for (int nt = 0; nt < 8; ++nt) {
    float bv = b2[nt * 16 + cit];
    f32x8 c;
#pragma unroll
    for (int r = 0; r < 8; ++r) c[r] = bv;
    acc2[nt] = c;
  }
#pragma unroll
  for (int kt = 0; kt < 4; ++kt) {
    const unsigned short* ap = my + cit * D + kt * 32 + ksel;
    bf16x16 a = load_a_frag(ap, ap + 16);
#pragma unroll
    for (int nt = 0; nt < 8; ++nt) {
      bf16x16 b = *(const bf16x16*)(w2 + (((kt << 3) + nt) * 32 + lane) * 16);
      acc2[nt] = wmma_bf16(a, b, acc2[nt]);
    }
  }

  // ---- residual + store
#pragma unroll
  for (int nt = 0; nt < 8; ++nt) {
    int c = nt * 16 + cit;
#pragma unroll
    for (int r = 0; r < 8; ++r) {
      long long n = n0 + mbase + r;
      if (n < N_) out[n * D + c] = acc2[nt][r] + x[n * D + c];
    }
  }
}

// ---------------------------------------------------------------------------
extern "C" void kernel_launch(void* const* d_in, const int* in_sizes, int n_in,
                              void* d_out, int out_size, void* d_ws,
                              size_t ws_size, hipStream_t stream) {
  (void)n_in; (void)out_size; (void)ws_size;
  const float* x   = (const float*)d_in[0];
  const int*   ei  = (const int*)d_in[1];
  const float* em  = (const float*)d_in[2];
  const float* ew1 = (const float*)d_in[3];
  const float* eb1 = (const float*)d_in[4];
  const float* ew2 = (const float*)d_in[5];
  const float* eb2 = (const float*)d_in[6];
  const float* nw1 = (const float*)d_in[7];
  const float* nb1 = (const float*)d_in[8];
  const float* nw2 = (const float*)d_in[9];
  const float* nb2 = (const float*)d_in[10];

  const int N = in_sizes[0] / D;
  const int E = in_sizes[1] / 2;

  // workspace layout
  char* ws = (char*)d_ws;
  size_t off = 0;
  auto alloc = [&](size_t bytes) -> void* {
    void* p = ws + off;
    off = (off + bytes + 255) & ~(size_t)255;
    return p;
  };
  unsigned short* xb    = (unsigned short*)alloc((size_t)N * D * 2);
  float*          agg   = (float*)alloc((size_t)N * D * 4);
  unsigned short* pw_e1 = (unsigned short*)alloc((size_t)2 * D * D * 2);
  unsigned short* pw_e2 = (unsigned short*)alloc((size_t)D * D * 2);
  unsigned short* pw_n1 = (unsigned short*)alloc((size_t)2 * D * D * 2);
  unsigned short* pw_n2 = (unsigned short*)alloc((size_t)D * D * 2);

  // 1) pack weights into WMMA B-fragment layout
  pack_w_kernel<<<(2 * D * D + 255) / 256, 256, 0, stream>>>(ew1, pw_e1, 8);
  pack_w_kernel<<<(D * D + 255) / 256, 256, 0, stream>>>(ew2, pw_e2, 4);
  pack_w_kernel<<<(2 * D * D + 255) / 256, 256, 0, stream>>>(nw1, pw_n1, 8);
  pack_w_kernel<<<(D * D + 255) / 256, 256, 0, stream>>>(nw2, pw_n2, 4);

  // 2) x -> bf16, agg -> 0
  long long total = (long long)N * D;
  cvt_zero_kernel<<<(int)((total + 255) / 256), 256, 0, stream>>>(x, xb, agg,
                                                                  total);

  float* out_nodes = (float*)d_out;
  float* out_edges = out_nodes + (size_t)N * D;

  // 3) edge MLP + scatter-add
  egcl_edge_kernel<<<(E + 63) / 64, 128, 0, stream>>>(
      ei, em, xb, pw_e1, eb1, pw_e2, eb2, out_edges, agg, E);

  // 4) node MLP + residual
  egcl_node_kernel<<<(N + 63) / 64, 128, 0, stream>>>(
      x, xb, agg, pw_n1, nb1, pw_n2, nb2, out_nodes, N);
}
